// Aggregator_21801253994883
// MI455X (gfx1250) — matrix operand order repaired and checked by reading, not compiled, not run on hardware
//
#include <hip/hip_runtime.h>
#include <hip/hip_bf16.h>

typedef __attribute__((ext_vector_type(16))) __bf16 v16bf;
typedef __attribute__((ext_vector_type(8)))  __bf16 v8bf;
typedef __attribute__((ext_vector_type(8)))  float  v8f;
typedef __attribute__((ext_vector_type(4)))  float  f32x4;
typedef __attribute__((ext_vector_type(4)))  int    v4i;

#define GLOBAL_AS __attribute__((address_space(1)))
#define SHARED_AS __attribute__((address_space(3)))

#define NTOT  500000
#define BSZ   128
#define SSZ   128
#define DSZ   128
#define CHUNK 32          // K of one bf16 WMMA step; nodes per chunk
#define NCHUNK (NTOT / CHUNK)

__global__ void zero_kernel(float* __restrict__ out, int n) {
    int i = blockIdx.x * blockDim.x + threadIdx.x;
    if (i < n) out[i] = 0.0f;
}

__global__ __launch_bounds__(256, 1)
void agg_kernel(const float* __restrict__ nodes,
                const int*   __restrict__ masks,
                const float* __restrict__ Wt,
                const float* __restrict__ bt,
                const float* __restrict__ Wg,
                const float* __restrict__ bg,
                float* __restrict__ out)
{
    __shared__ __bf16 nodesA[CHUNK][SSZ];   //  8 KB  node chunk, bf16, [node][feature]
    __shared__ __bf16 xB[CHUNK][DSZ];       //  8 KB  gated data,  [node][d]
    __shared__ int    maskI[BSZ][CHUNK];    // 16 KB  raw mask chunk via async DMA

    const int tid   = threadIdx.x;
    const int wave  = tid >> 5;      // 0..7
    const int lane  = tid & 31;
    const int lhalf = lane >> 4;     // 0/1 (lane half, drives K split in fragments)
    const int l15   = lane & 15;

    const int nt = wave;             // this wave's D-tile for the feature GEMMs

    // ---- preload this wave's Wt/Wg B-fragments (bf16) into registers -------
    // B-matrix 32x16 bf16, contiguous k order chosen here: lanes 0-15 hold K=0..15 (col = lane),
    // lanes 16-31 hold K=16..31 (col = lane-16); VGPR v packs K = base+2v, +1.
    v16bf wtB[4], wgB[4];
    {
        const int d = nt * 16 + l15;
        #pragma unroll
        for (int k = 0; k < 4; ++k) {
            const int sbase = k * 32 + lhalf * 16;
            #pragma unroll
            for (int v = 0; v < 8; ++v) {
                const int s = sbase + 2 * v;
                wtB[k][2*v]   = (__bf16)Wt[s * DSZ + d];
                wtB[k][2*v+1] = (__bf16)Wt[(s + 1) * DSZ + d];
                wgB[k][2*v]   = (__bf16)Wg[s * DSZ + d];
                wgB[k][2*v+1] = (__bf16)Wg[(s + 1) * DSZ + d];
            }
        }
    }
    const float btv = bt[nt * 16 + l15];
    const float bgv = bg[nt * 16 + l15];

    // ---- persistent pooling accumulators: wave owns B-rows [16*wave, 16*wave+16)
    v8f pacc[8];
    #pragma unroll
    for (int i = 0; i < 8; ++i) pacc[i] = (v8f){0.f,0.f,0.f,0.f,0.f,0.f,0.f,0.f};

    for (int c = blockIdx.x; c < NCHUNK; c += gridDim.x) {
        const long n0 = (long)c * CHUNK;

        // ---- kick mask chunk DMA: global -> LDS, no VGPR round-trip --------
        // 128 rows x 32 int32 = 16 KB; each lane moves one b128 (4 ints),
        // lanes 0..7 cover one 128B row => fully coalesced. Tracked on ASYNCcnt.
        {
            const int cc = (tid & 7) * 4;
            #pragma unroll
            for (int it = 0; it < 4; ++it) {
                const int b = (tid >> 3) + 32 * it;
                __builtin_amdgcn_global_load_async_to_lds_b128(
                    (GLOBAL_AS v4i*)(masks + (long)b * NTOT + n0 + cc),
                    (SHARED_AS v4i*)&maskI[b][cc],
                    0, 0);
            }
        }

        // ---- stage node chunk (coalesced, non-temporal, packed bf16 stores) -
        {
            const int node = tid >> 3;          // 32 nodes, 8 threads each
            const int seg  = tid & 7;           // 16 floats per thread
            const float* src = nodes + (n0 + node) * SSZ + seg * 16;
            __builtin_prefetch(src + (long)gridDim.x * CHUNK * SSZ, 0, 0);
            #pragma unroll
            for (int h = 0; h < 2; ++h) {
                f32x4 a0 = __builtin_nontemporal_load((const f32x4*)(src + 8 * h));
                f32x4 a1 = __builtin_nontemporal_load((const f32x4*)(src + 8 * h + 4));
                v8bf p;
                p[0] = (__bf16)a0.x; p[1] = (__bf16)a0.y;
                p[2] = (__bf16)a0.z; p[3] = (__bf16)a0.w;
                p[4] = (__bf16)a1.x; p[5] = (__bf16)a1.y;
                p[6] = (__bf16)a1.z; p[7] = (__bf16)a1.w;
                *(v8bf*)&nodesA[node][seg * 16 + 8 * h] = p;   // ds_store_b128
            }
        }
        __syncthreads();   // nodesA visible to all waves

        // ---- feature GEMMs: y = nodes_blk @ {Wt,Wg} for this wave's D-tile --
        #pragma unroll
        for (int m = 0; m < 2; ++m) {           // two 16-row node tiles
            v8f accT = (v8f){0.f,0.f,0.f,0.f,0.f,0.f,0.f,0.f};
            v8f accG = (v8f){0.f,0.f,0.f,0.f,0.f,0.f,0.f,0.f};
            #pragma unroll
            for (int k = 0; k < 4; ++k) {
                // A-matrix 16x32 bf16 fragment (row = m*16 + l15)
                v16bf a;
                const int row = m * 16 + l15;
                #pragma unroll
                for (int v = 0; v < 8; ++v) {
                    const int kk = k * 32 + lhalf * 16 + 2 * v;
                    a[2*v]   = nodesA[row][kk];
                    a[2*v+1] = nodesA[row][kk + 1];
                }
                accT = __builtin_amdgcn_wmma_f32_16x16x32_bf16(
                           false, a, false, wtB[k], (short)0, accT, false, false);
                accG = __builtin_amdgcn_wmma_f32_16x16x32_bf16(
                           false, a, false, wgB[k], (short)0, accG, false, false);
            }
            // fused bias + sigmoid gate, write gated data to LDS (bf16)
            // C/D layout: elem r in lane L -> row = r + 8*(L>=16), col = L&15
            #pragma unroll
            for (int r = 0; r < 8; ++r) {
                const float dv = accT[r] + btv;
                const float gv = accG[r] + bgv;
                const float sg = 1.0f / (1.0f + __expf(-gv));
                const int node = m * 16 + r + lhalf * 8;
                xB[node][nt * 16 + l15] = (__bf16)(dv * sg);
            }
        }

        // mask DMA must have landed before pooling reads; each wave drains its
        // own ASYNCcnt, then the barrier orders it block-wide (also covers xB).
        __builtin_amdgcn_s_wait_asynccnt(0);
        __syncthreads();

        // ---- pooling GEMM: out[16 rows of B, :] += mask_tile @ xB ----------
        v16bf am;
        {
            const int brow = wave * 16 + l15;
            #pragma unroll
            for (int v = 0; v < 8; ++v) {
                const int kk = lhalf * 16 + 2 * v;
                am[2*v]   = (__bf16)(float)maskI[brow][kk];
                am[2*v+1] = (__bf16)(float)maskI[brow][kk + 1];
            }
        }
        #pragma unroll
        for (int dt = 0; dt < 8; ++dt) {
            v16bf bx;
            const int col = dt * 16 + l15;
            #pragma unroll
            for (int v = 0; v < 8; ++v) {
                const int kk = lhalf * 16 + 2 * v;
                bx[2*v]   = xB[kk][col];
                bx[2*v+1] = xB[kk + 1][col];
            }
            pacc[dt] = __builtin_amdgcn_wmma_f32_16x16x32_bf16(
                           false, am, false, bx, (short)0, pacc[dt], false, false);
        }
        __syncthreads();   // protect nodesA/xB/maskI against next iteration
    }

    // ---- flush persistent accumulators (512 partial sums per output elem) --
    #pragma unroll
    for (int dt = 0; dt < 8; ++dt) {
        #pragma unroll
        for (int r = 0; r < 8; ++r) {
            const int brow = wave * 16 + r + lhalf * 8;
            const int col  = dt * 16 + l15;
            atomicAdd(&out[brow * DSZ + col], pacc[dt][r]);
        }
    }
}

extern "C" void kernel_launch(void* const* d_in, const int* in_sizes, int n_in,
                              void* d_out, int out_size, void* d_ws, size_t ws_size,
                              hipStream_t stream) {
    const float* nodes = (const float*)d_in[0];
    const int*   masks = (const int*)  d_in[1];
    const float* Wt    = (const float*)d_in[2];
    const float* bt    = (const float*)d_in[3];
    const float* Wg    = (const float*)d_in[4];
    const float* bg    = (const float*)d_in[5];
    float* out = (float*)d_out;

    zero_kernel<<<(BSZ * DSZ + 255) / 256, 256, 0, stream>>>(out, BSZ * DSZ);
    agg_kernel<<<512, 256, 0, stream>>>(nodes, masks, Wt, bt, Wg, bg, out);
}
